// Decoder_12850542150578
// MI455X (gfx1250) — matrix operand order repaired and checked by reading, not compiled, not run on hardware
//
#include <hip/hip_runtime.h>
#include <hip/hip_bf16.h>
#include <math.h>

// Problem constants (from reference)
#define BB 32
#define TT 64
#define SS 64
#define VV 32000
#define EE 512
#define HH 512

typedef __attribute__((ext_vector_type(8)))  float  v8f;
typedef __attribute__((ext_vector_type(8)))  __bf16 bf16x8;
typedef __attribute__((ext_vector_type(16))) __bf16 bf16x16;

// ---------------------------------------------------------------------------
// WMMA tile loaders (layouts per CDNA5 ISA §7.12.2, 16-bit A 16x32 / B 32x16)
// ---------------------------------------------------------------------------
// A tile (16x32, MxK): lane<16 -> M=lane, elements 0..7 = K k0..k0+7,
// elements 8..15 = K k0+16..k0+23 ; lanes 16..31 use K offset +8.
__device__ inline bf16x16 ld_a_tile(const __bf16* __restrict__ A, int lda,
                                    int row0, int k0, int lane) {
  int m  = row0 + (lane & 15);
  int kb = k0 + ((lane >> 4) << 3);
  const __bf16* p = A + (size_t)m * lda + kb;
  bf16x8 lo = *reinterpret_cast<const bf16x8*>(p);
  bf16x8 hi = *reinterpret_cast<const bf16x8*>(p + 16);
  bf16x16 v;
#pragma unroll
  for (int i = 0; i < 8; ++i) { v[i] = lo[i]; v[i + 8] = hi[i]; }
  return v;
}

// B tile (32x16, KxN) for C = A @ W^T with W row-major [N,K]:
// lane holds column n = n0 + (lane&15); elements 0..7 = K kb..kb+7, 8..15 =
// K kb+16..kb+23 with kb = k0 + 8*(lane>=16) (matches A tile order).
__device__ inline bf16x16 ld_b_tile(const __bf16* __restrict__ W, int ldw,
                                    int n0, int k0, int lane) {
  int n  = n0 + (lane & 15);
  int kk = k0 + ((lane >> 4) << 3);
  const __bf16* p = W + (size_t)n * ldw + kk;
  bf16x8 lo = *reinterpret_cast<const bf16x8*>(p);
  bf16x8 hi = *reinterpret_cast<const bf16x8*>(p + 16);
  bf16x16 v;
#pragma unroll
  for (int i = 0; i < 8; ++i) { v[i] = lo[i]; v[i + 8] = hi[i]; }
  return v;
}

// ---------------------------------------------------------------------------
// Dual-source WMMA GEMM: C[M,N] = A1@W1^T (+ A2@W2^T) (+ bias1 + bias2)
// Block = 256 threads = 8 waves; block tile 32(M) x 128(N); wave tile 32x16.
// grid.x = N/128, grid.y = M/32. C is fp32 with row stride ldc.
// ---------------------------------------------------------------------------
__global__ void __launch_bounds__(256)
wmma_gemm_dual(const __bf16* __restrict__ A1, const __bf16* __restrict__ W1, int K1,
               const __bf16* __restrict__ A2, const __bf16* __restrict__ W2, int K2,
               const float* __restrict__ bias1, const float* __restrict__ bias2,
               float* __restrict__ C, int ldc) {
  const int lane = threadIdx.x & 31;
  const int wave = threadIdx.x >> 5;
  const int row0 = blockIdx.y * 32;
  const int col  = blockIdx.x * 128 + wave * 16;

  v8f acc0 = {};
  v8f acc1 = {};

  for (int k0 = 0; k0 < K1; k0 += 32) {
    bf16x16 a0 = ld_a_tile(A1, K1, row0,      k0, lane);
    bf16x16 a1 = ld_a_tile(A1, K1, row0 + 16, k0, lane);
    bf16x16 b  = ld_b_tile(W1, K1, col,       k0, lane);
    acc0 = __builtin_amdgcn_wmma_f32_16x16x32_bf16(false, a0, false, b,
                                                   (short)0, acc0, false, false);
    acc1 = __builtin_amdgcn_wmma_f32_16x16x32_bf16(false, a1, false, b,
                                                   (short)0, acc1, false, false);
  }
  if (A2) {
    for (int k0 = 0; k0 < K2; k0 += 32) {
      bf16x16 a0 = ld_a_tile(A2, K2, row0,      k0, lane);
      bf16x16 a1 = ld_a_tile(A2, K2, row0 + 16, k0, lane);
      bf16x16 b  = ld_b_tile(W2, K2, col,       k0, lane);
      acc0 = __builtin_amdgcn_wmma_f32_16x16x32_bf16(false, a0, false, b,
                                                     (short)0, acc0, false, false);
      acc1 = __builtin_amdgcn_wmma_f32_16x16x32_bf16(false, a1, false, b,
                                                     (short)0, acc1, false, false);
    }
  }

  const int n = col + (lane & 15);
  float bs = 0.f;
  if (bias1) bs += bias1[n];
  if (bias2) bs += bias2[n];
  // C/D layout: VGPR r -> M = row0 + r + 8*(lane>=16), N = lane&15
  const int mb = row0 + ((lane >> 4) << 3);
#pragma unroll
  for (int r = 0; r < 8; ++r) {
    C[(size_t)(mb + r) * ldc + n]      = acc0[r] + bs;
    C[(size_t)(mb + 16 + r) * ldc + n] = acc1[r] + bs;
  }
}

// ---------------------------------------------------------------------------
// LSTM gate activations: gates [B,4H] (i,f,g,o), c updated in place,
// h written fp32 + bf16 (optionally into a second strided bf16 buffer).
// ---------------------------------------------------------------------------
__global__ void __launch_bounds__(256)
lstm_act(const float* __restrict__ gates, float* __restrict__ c_state,
         float* __restrict__ h_state_f, __bf16* __restrict__ h_bf0,
         __bf16* __restrict__ h_bf1, int h_bf1_stride) {
  int idx = blockIdx.x * blockDim.x + threadIdx.x;
  if (idx >= BB * HH) return;
  int b = idx >> 9, j = idx & (HH - 1);
  const float* g = gates + (size_t)b * 4 * HH;
  float gi = g[j], gf = g[HH + j], gg = g[2 * HH + j], go = g[3 * HH + j];
  float si = 1.f / (1.f + __expf(-gi));
  float sf = 1.f / (1.f + __expf(-gf));
  float so = 1.f / (1.f + __expf(-go));
  float c  = sf * c_state[idx] + si * tanhf(gg);
  float h  = so * tanhf(c);
  c_state[idx]   = c;
  h_state_f[idx] = h;
  h_bf0[(size_t)b * HH + j] = (__bf16)h;
  if (h_bf1) h_bf1[(size_t)b * h_bf1_stride + j] = (__bf16)h;
}

// ---------------------------------------------------------------------------
// Per-step attention: scores -> softmax -> context (bf16 into both GEMM
// input buffers), plus copy of this step's embedding into inp_bf.
// One block per batch element, 64 threads.
// ---------------------------------------------------------------------------
__global__ void __launch_bounds__(64)
attention_step(const float* __restrict__ enc_proj, const float* __restrict__ enc_out,
               const float* __restrict__ h1_f, const __bf16* __restrict__ embedded,
               int t, __bf16* __restrict__ inp_bf, __bf16* __restrict__ outin_bf) {
  __shared__ float sc[SS];
  __shared__ float red[SS];
  const int tid = threadIdx.x;
  const int b   = blockIdx.x;

  const float* ep = enc_proj + ((size_t)b * SS + tid) * HH;
  const float* hv = h1_f + (size_t)b * HH;
  float dot = 0.f;
  for (int h = 0; h < HH; ++h) dot = fmaf(ep[h], hv[h], dot);
  sc[tid] = dot;
  red[tid] = dot;
  __syncthreads();
  for (int off = 32; off > 0; off >>= 1) {
    if (tid < off) red[tid] = fmaxf(red[tid], red[tid + off]);
    __syncthreads();
  }
  float mx = red[0];
  __syncthreads();
  float e = __expf(sc[tid] - mx);
  sc[tid] = e;
  red[tid] = e;
  __syncthreads();
  for (int off = 32; off > 0; off >>= 1) {
    if (tid < off) red[tid] += red[tid + off];
    __syncthreads();
  }
  float inv = 1.f / red[0];

  for (int h = tid; h < HH; h += 64) {
    const float* eb = enc_out + (size_t)b * SS * HH + h;
    float cx = 0.f;
    for (int s = 0; s < SS; ++s) cx = fmaf(sc[s], eb[(size_t)s * HH], cx);
    cx *= inv;
    __bf16 cb = (__bf16)cx;
    inp_bf[(size_t)b * (EE + HH) + EE + h]   = cb;
    outin_bf[(size_t)b * (HH + HH) + HH + h] = cb;
  }
  const __bf16* er = embedded + ((size_t)b * TT + t) * EE;
  for (int j = tid; j < EE; j += 64) inp_bf[(size_t)b * (EE + HH) + j] = er[j];
}

// ---------------------------------------------------------------------------
// One-time prep kernels
// ---------------------------------------------------------------------------
__global__ void f2bf(const float* __restrict__ s, __bf16* __restrict__ d, size_t n) {
  size_t i = (size_t)blockIdx.x * blockDim.x + threadIdx.x;
  size_t stride = (size_t)gridDim.x * blockDim.x;
  for (; i < n; i += stride) d[i] = (__bf16)s[i];
}

__global__ void transpose_f2bf(const float* __restrict__ s, __bf16* __restrict__ d,
                               int R, int Cc) {
  int i = blockIdx.x * blockDim.x + threadIdx.x;
  if (i >= R * Cc) return;
  int r = i / Cc, c = i - r * Cc;
  d[(size_t)c * R + r] = (__bf16)s[i];
}

__global__ void gather_emb(const int* __restrict__ tgt, const float* __restrict__ emb,
                           __bf16* __restrict__ out) {
  size_t i = (size_t)blockIdx.x * blockDim.x + threadIdx.x;
  if (i >= (size_t)BB * TT * EE) return;
  size_t row = i >> 9;           // E = 512
  int e = (int)(i & (EE - 1));
  int tok = tgt[row];
  out[i] = (tok == 0) ? (__bf16)0.f : (__bf16)emb[(size_t)tok * EE + e];
}

__global__ void init_state(const float* __restrict__ hidden, const float* __restrict__ cellin,
                           float* __restrict__ h0, float* __restrict__ h1,
                           float* __restrict__ c0, float* __restrict__ c1,
                           __bf16* __restrict__ h0b, __bf16* __restrict__ h1b) {
  int idx = blockIdx.x * blockDim.x + threadIdx.x;
  if (idx >= BB * HH) return;
  float a = hidden[idx], bv = hidden[BB * HH + idx];
  h0[idx] = a;  h1[idx] = bv;
  c0[idx] = cellin[idx];
  c1[idx] = cellin[BB * HH + idx];
  h0b[idx] = (__bf16)a;
  h1b[idx] = (__bf16)bv;
}

__global__ void write_state(const float* __restrict__ h0, const float* __restrict__ h1,
                            const float* __restrict__ c0, const float* __restrict__ c1,
                            float* __restrict__ tail) {
  int idx = blockIdx.x * blockDim.x + threadIdx.x;
  if (idx >= BB * HH) return;
  tail[idx]               = h0[idx];
  tail[BB * HH + idx]     = h1[idx];
  tail[2 * BB * HH + idx] = c0[idx];
  tail[3 * BB * HH + idx] = c1[idx];
}

// ---------------------------------------------------------------------------
extern "C" void kernel_launch(void* const* d_in, const int* in_sizes, int n_in,
                              void* d_out, int out_size, void* d_ws, size_t ws_size,
                              hipStream_t stream) {
  (void)in_sizes; (void)n_in; (void)out_size; (void)ws_size;

  const int*   tgt     = (const int*)d_in[0];
  const float* enc_out = (const float*)d_in[1];
  const float* hidden  = (const float*)d_in[2];
  const float* cellin  = (const float*)d_in[3];
  const float* emb     = (const float*)d_in[4];
  const float* W_a     = (const float*)d_in[5];
  const float* W_ih0   = (const float*)d_in[6];
  const float* W_hh0   = (const float*)d_in[7];
  const float* b_ih0   = (const float*)d_in[8];
  const float* b_hh0   = (const float*)d_in[9];
  const float* W_ih1   = (const float*)d_in[10];
  const float* W_hh1   = (const float*)d_in[11];
  const float* b_ih1   = (const float*)d_in[12];
  const float* b_hh1   = (const float*)d_in[13];
  const float* W_out   = (const float*)d_in[14];
  const float* b_out   = (const float*)d_in[15];
  float* outs = (float*)d_out;

  // --- workspace carve-out ---
  char* ws = (char*)d_ws;
  size_t off = 0;
  auto alloc = [&](size_t bytes) -> void* {
    off = (off + 255) & ~(size_t)255;
    void* p = ws + off;
    off += bytes;
    return p;
  };
  __bf16* wout_bf = (__bf16*)alloc((size_t)VV * 2 * HH * 2);        // [V, 2H]
  __bf16* wih0_bf = (__bf16*)alloc((size_t)4 * HH * (EE + HH) * 2); // [4H, E+H]
  __bf16* whh0_bf = (__bf16*)alloc((size_t)4 * HH * HH * 2);
  __bf16* wih1_bf = (__bf16*)alloc((size_t)4 * HH * HH * 2);
  __bf16* whh1_bf = (__bf16*)alloc((size_t)4 * HH * HH * 2);
  __bf16* waT_bf  = (__bf16*)alloc((size_t)HH * HH * 2);            // W_a^T
  __bf16* enc_bf  = (__bf16*)alloc((size_t)BB * SS * HH * 2);
  __bf16* emb_bf  = (__bf16*)alloc((size_t)BB * TT * EE * 2);
  float*  enc_prj = (float*)alloc((size_t)BB * SS * HH * 4);
  float*  h0_f    = (float*)alloc((size_t)BB * HH * 4);
  float*  h1_f    = (float*)alloc((size_t)BB * HH * 4);
  float*  c0_f    = (float*)alloc((size_t)BB * HH * 4);
  float*  c1_f    = (float*)alloc((size_t)BB * HH * 4);
  __bf16* h0_bf   = (__bf16*)alloc((size_t)BB * HH * 2);
  __bf16* h1_bf   = (__bf16*)alloc((size_t)BB * HH * 2);
  __bf16* inp_bf  = (__bf16*)alloc((size_t)BB * (EE + HH) * 2);     // [emb_t | ctx]
  __bf16* oin_bf  = (__bf16*)alloc((size_t)BB * (HH + HH) * 2);     // [h1 | ctx]
  float*  gates_f = (float*)alloc((size_t)BB * 4 * HH * 4);

  // --- one-time prep ---
  auto cvt = [&](const float* s, __bf16* d, size_t n) {
    int blocks = (int)((n + 255) / 256);
    if (blocks > 8192) blocks = 8192;
    hipLaunchKernelGGL(f2bf, dim3(blocks), dim3(256), 0, stream, s, d, n);
  };
  cvt(W_out, wout_bf, (size_t)VV * 2 * HH);
  cvt(W_ih0, wih0_bf, (size_t)4 * HH * (EE + HH));
  cvt(W_hh0, whh0_bf, (size_t)4 * HH * HH);
  cvt(W_ih1, wih1_bf, (size_t)4 * HH * HH);
  cvt(W_hh1, whh1_bf, (size_t)4 * HH * HH);
  cvt(enc_out, enc_bf, (size_t)BB * SS * HH);
  hipLaunchKernelGGL(transpose_f2bf, dim3((HH * HH + 255) / 256), dim3(256), 0, stream,
                     W_a, waT_bf, HH, HH);
  hipLaunchKernelGGL(gather_emb, dim3((BB * TT * EE + 255) / 256), dim3(256), 0, stream,
                     tgt, emb, emb_bf);
  hipLaunchKernelGGL(init_state, dim3((BB * HH + 255) / 256), dim3(256), 0, stream,
                     hidden, cellin, h0_f, h1_f, c0_f, c1_f, h0_bf, h1_bf);

  // enc_proj = enc @ W_a  : M = B*S = 2048, N = H = 512, K = H = 512
  hipLaunchKernelGGL(wmma_gemm_dual, dim3(HH / 128, (BB * SS) / 32), dim3(256), 0, stream,
                     enc_bf, waT_bf, HH,
                     (const __bf16*)nullptr, (const __bf16*)nullptr, 0,
                     (const float*)nullptr, (const float*)nullptr,
                     enc_prj, HH);

  // --- sequential scan over T timesteps ---
  for (int t = 0; t < TT; ++t) {
    // attention + assemble inp_bf = [emb_t | context], oin_bf ctx half
    hipLaunchKernelGGL(attention_step, dim3(BB), dim3(64), 0, stream,
                       enc_prj, enc_out, h1_f, emb_bf, t, inp_bf, oin_bf);

    // layer-0 gates = inp @ W_ih0^T + h0 @ W_hh0^T + b_ih0 + b_hh0
    hipLaunchKernelGGL(wmma_gemm_dual, dim3((4 * HH) / 128, 1), dim3(256), 0, stream,
                       inp_bf, wih0_bf, EE + HH,
                       h0_bf, whh0_bf, HH,
                       b_ih0, b_hh0, gates_f, 4 * HH);
    hipLaunchKernelGGL(lstm_act, dim3((BB * HH + 255) / 256), dim3(256), 0, stream,
                       gates_f, c0_f, h0_f, h0_bf, (__bf16*)nullptr, 0);

    // layer-1 gates = h0_new @ W_ih1^T + h1_prev @ W_hh1^T + b_ih1 + b_hh1
    hipLaunchKernelGGL(wmma_gemm_dual, dim3((4 * HH) / 128, 1), dim3(256), 0, stream,
                       h0_bf, wih1_bf, HH,
                       h1_bf, whh1_bf, HH,
                       b_ih1, b_hh1, gates_f, 4 * HH);
    // h1 bf16 also lands in oin_bf[:, 0:H] (stride 2H)
    hipLaunchKernelGGL(lstm_act, dim3((BB * HH + 255) / 256), dim3(256), 0, stream,
                       gates_f, c1_f, h1_f, h1_bf, oin_bf, 2 * HH);

    // vocab projection: [B, 2H] @ W_out^T + b_out -> d_out[:, t, :]
    hipLaunchKernelGGL(wmma_gemm_dual, dim3(VV / 128, 1), dim3(256), 0, stream,
                       oin_bf, wout_bf, 2 * HH,
                       (const __bf16*)nullptr, (const __bf16*)nullptr, 0,
                       b_out, (const float*)nullptr,
                       outs + (size_t)t * VV, TT * VV);
  }

  // final h_f / c_f into tail of d_out
  hipLaunchKernelGGL(write_state, dim3((BB * HH + 255) / 256), dim3(256), 0, stream,
                     h0_f, h1_f, c0_f, c1_f, outs + (size_t)BB * TT * VV);
}
